// SimpleNoisyTopKGatingNetwork_85718957294268
// MI455X (gfx1250) — matrix operand order repaired and checked by reading, not compiled, not run on hardware
//
#include <hip/hip_runtime.h>
#include <hip/hip_bf16.h>
#include <math.h>

// ---------------------------------------------------------------------------
// MoE noisy top-2 gating: conv3x3(64->64) + ReLU + spatial mean -> gating.
// Conv implemented as implicit GEMM on v_wmma_f32_16x16x32_bf16 with a
// hi/lo bf16 split (3 WMMA products) for ~fp32 accuracy.
// Weight taps staged into LDS with GLOBAL_LOAD_ASYNC_TO_LDS_B128 (ASYNCcnt).
// ---------------------------------------------------------------------------

typedef __attribute__((ext_vector_type(16))) __bf16 v16bf;
typedef __attribute__((ext_vector_type(8)))  float  v8f;

union FragBF {
    v16bf v;
    uint4 q[2];
};

__device__ __forceinline__ unsigned short f2bf(float f) {
    unsigned u = __float_as_uint(f);
    unsigned r = 0x7FFFu + ((u >> 16) & 1u);   // round-to-nearest-even
    return (unsigned short)((u + r) >> 16);
}
__device__ __forceinline__ float bf2f(unsigned short h) {
    return __uint_as_float(((unsigned)h) << 16);
}

// Async 16-byte global -> LDS copy (GV mode, null saddr). Tracked by ASYNCcnt.
__device__ __forceinline__ void async_copy16(unsigned ldsOff, const void* gptr) {
    unsigned long long ga = (unsigned long long)(uintptr_t)gptr;
    asm volatile("global_load_async_to_lds_b128 %0, %1, off"
                 :: "v"(ldsOff), "v"(ga) : "memory");
}
__device__ __forceinline__ void wait_async0() {
    asm volatile("s_wait_asynccnt 0x0" ::: "memory");
}

#define IMG_C   64
#define IMG_HW  64
#define OUT_C   64
#define ROWS    2          // output rows per block
#define RB      32         // row-blocks per image (64 / ROWS)
#define XCOLP   66         // 64 cols + 1 halo each side
#define XCHP    72         // channel pad (multiple of 8, de-phases banks)
#define WOCP    72         // out-channel stride pad in weight LDS

// ---------------------------------------------------------------------------
// Kernel 1: transform conv_w (OIHW, fp32) -> per-tap [tap][o][c] bf16 hi/lo
// ---------------------------------------------------------------------------
__global__ void prep_w_kernel(const float* __restrict__ conv_w,
                              unsigned short* __restrict__ whiT,
                              unsigned short* __restrict__ wloT) {
    int i = blockIdx.x * 256 + threadIdx.x;       // [tap][o][c]
    if (i >= 9 * OUT_C * IMG_C) return;
    int tap = i >> 12;
    int oc  = i & 4095;
    int o   = oc >> 6;
    int c   = oc & 63;
    int kh  = tap / 3, kw = tap % 3;
    float v = conv_w[((o * IMG_C + c) * 3 + kh) * 3 + kw];
    unsigned short h = f2bf(v);
    whiT[i] = h;
    wloT[i] = f2bf(v - bf2f(h));
}

// ---------------------------------------------------------------------------
// Kernel 2: conv3x3 + bias + relu + partial spatial sum, WMMA bf16 hi/lo.
// Grid: (RB, B). Block: 256 threads = 8 waves. Each block: 2 rows x 64 cols
// x 64 outch; writes partial[b][rb][64].
// ---------------------------------------------------------------------------
__global__ __launch_bounds__(256) void conv_feat_kernel(
        const float* __restrict__ x,
        const unsigned short* __restrict__ whiT,
        const unsigned short* __restrict__ wloT,
        const float* __restrict__ conv_b,
        float* __restrict__ partial) {
    __shared__ __align__(16) unsigned short xhi[ROWS * XCOLP * XCHP];
    __shared__ __align__(16) unsigned short xlo[ROWS * XCOLP * XCHP];
    __shared__ __align__(16) unsigned short whi[OUT_C * WOCP];
    __shared__ __align__(16) unsigned short wlo[OUT_C * WOCP];
    __shared__ float redf[8 * OUT_C];

    const int b    = blockIdx.y;
    const int rb   = blockIdx.x;
    const int r0   = rb * ROWS;
    const int tid  = threadIdx.x;
    const int wv   = tid >> 5;        // wave 0..7
    const int lane = tid & 31;
    const int m    = lane & 15;
    const int hiHalf = lane >> 4;     // 0 or 1
    const int rowl    = wv >> 2;      // wave's local output row (0..1)
    const int colBase = (wv & 3) * 16;

    const unsigned whiBase = (unsigned)(uintptr_t)&whi[0];
    const unsigned wloBase = (unsigned)(uintptr_t)&wlo[0];

    v8f acc[4];
    for (int n = 0; n < 4; ++n)
        for (int i = 0; i < 8; ++i) acc[n][i] = 0.0f;

    // Zero the column-halo (colp 0 and 65) once; never overwritten below.
    for (int i = tid; i < ROWS * 2 * XCHP; i += 256) {
        int slot = i / (2 * XCHP);
        int rem  = i % (2 * XCHP);
        int colp = (rem >= XCHP) ? (XCOLP - 1) : 0;
        int ch   = rem % XCHP;
        int idx  = (slot * XCOLP + colp) * XCHP + ch;
        xhi[idx] = 0; xlo[idx] = 0;
    }

    for (int kh = 0; kh < 3; ++kh) {
        __syncthreads();   // previous tap compute done before x reload
        // Stage the 2-row input band (rows r0+kh-1 .. r0+kh) as bf16 hi/lo.
        for (int i = tid; i < ROWS * IMG_HW * IMG_C; i += 256) {
            int c    = i >> 7;          // channel
            int rem  = i & 127;
            int slot = rem >> 6;        // local row slot
            int col  = rem & 63;
            int g    = r0 + slot + kh - 1;
            float v  = 0.0f;
            if ((unsigned)g < (unsigned)IMG_HW)
                v = x[(((size_t)b * IMG_C + c) * IMG_HW + g) * IMG_HW + col];
            unsigned short h = f2bf(v);
            int idx = (slot * XCOLP + col + 1) * XCHP + c;
            xhi[idx] = h;
            xlo[idx] = f2bf(v - bf2f(h));
        }
        // Prefetch next kh's new image row (one 256B line per channel).
        if (kh < 2 && tid < IMG_C) {
            int g = r0 + kh + 1;
            if (g < IMG_HW)
                __builtin_prefetch(
                    &x[(((size_t)b * IMG_C + tid) * IMG_HW + g) * IMG_HW], 0, 3);
        }
        for (int kw = 0; kw < 3; ++kw) {
            __syncthreads();  // x stores visible / previous w-tap consumed
            // Async-DMA the (kh,kw) weight tap (bf16 hi/lo planes) into LDS.
            // 512 chunks of 16B per plane; per-lane LDS addr applies WOCP pad.
            {
                const int tap = kh * 3 + kw;
                #pragma unroll
                for (int i = tid; i < 512; i += 256) {
                    int o  = i >> 3;
                    int c8 = (i & 7) * 8;           // channel start of chunk
                    unsigned ldsO = (unsigned)((o * WOCP + c8) * 2);
                    size_t   gO   = (size_t)(tap * (OUT_C * IMG_C) + o * IMG_C + c8);
                    async_copy16(whiBase + ldsO, whiT + gO);
                    async_copy16(wloBase + ldsO, wloT + gO);
                }
                wait_async0();   // our async writes done before barrier signal
            }
            __syncthreads();

            #pragma unroll
            for (int k0 = 0; k0 < IMG_C; k0 += 32) {
                // A fragment (16x32 bf16): lane m = pixel colBase+m.
                // lanes 0-15: K k0+0..7 then k0+16..23 ; lanes 16-31: +8.
                FragBF ah, al;
                int axbase = (rowl * XCOLP + colBase + m + kw) * XCHP
                             + k0 + hiHalf * 8;
                ah.q[0] = *(const uint4*)&xhi[axbase];
                ah.q[1] = *(const uint4*)&xhi[axbase + 16];
                al.q[0] = *(const uint4*)&xlo[axbase];
                al.q[1] = *(const uint4*)&xlo[axbase + 16];
                #pragma unroll
                for (int n = 0; n < 4; ++n) {
                    // B fragment (32x16 bf16): lane = out-channel column,
                    // lanes 0-15: K 0..7 then 16..23 ; lanes 16-31: +8.
                    FragBF bh, bl;
                    int wbase = (n * 16 + m) * WOCP + k0 + hiHalf * 8;
                    bh.q[0] = *(const uint4*)&whi[wbase];
                    bh.q[1] = *(const uint4*)&whi[wbase + 16];
                    bl.q[0] = *(const uint4*)&wlo[wbase];
                    bl.q[1] = *(const uint4*)&wlo[wbase + 16];
                    acc[n] = __builtin_amdgcn_wmma_f32_16x16x32_bf16(
                        false, ah.v, false, bh.v, (short)0, acc[n], false, false);
                    acc[n] = __builtin_amdgcn_wmma_f32_16x16x32_bf16(
                        false, ah.v, false, bl.v, (short)0, acc[n], false, false);
                    acc[n] = __builtin_amdgcn_wmma_f32_16x16x32_bf16(
                        false, al.v, false, bh.v, (short)0, acc[n], false, false);
                }
            }
        }
    }

    // Epilogue: bias + relu + sum over this wave's 16 pixels per out-channel.
    #pragma unroll
    for (int n = 0; n < 4; ++n) {
        int o = n * 16 + m;              // every lane's 8 values share N = m
        float bias = conv_b[o];
        float s = 0.0f;
        #pragma unroll
        for (int i = 0; i < 8; ++i) {
            float v = acc[n][i] + bias;
            s += fmaxf(v, 0.0f);
        }
        s += __shfl_xor(s, 16, 32);      // combine M 0-7 with M 8-15 halves
        if (hiHalf == 0) redf[wv * OUT_C + o] = s;
    }
    __syncthreads();
    if (tid < OUT_C) {                   // deterministic cross-wave reduce
        float s = 0.0f;
        for (int w2 = 0; w2 < 8; ++w2) s += redf[w2 * OUT_C + tid];
        partial[((size_t)b * RB + rb) * OUT_C + tid] = s;
    }
}

// ---------------------------------------------------------------------------
// Kernel 3: gating. One block, 256 threads (one per batch row).
// ---------------------------------------------------------------------------
__global__ __launch_bounds__(256) void gate_kernel(
        const float* __restrict__ partial,
        const float* __restrict__ gate_w, const float* __restrict__ gate_b,
        const float* __restrict__ noise_w, const float* __restrict__ noise_b,
        const float* __restrict__ noise,
        float* __restrict__ out) {         // [0,4096) gates, [4096,4112) load
    __shared__ float gsh[256 * 16];
    const int b = threadIdx.x;

    float feat[64];
    for (int o = 0; o < 64; ++o) feat[o] = 0.0f;
    for (int r = 0; r < RB; ++r) {
        const float* p = partial + ((size_t)b * RB + r) * OUT_C;
        for (int o = 0; o < 64; ++o) feat[o] += p[o];
    }
    const float inv = 1.0f / 4096.0f;
    for (int o = 0; o < 64; ++o) feat[o] *= inv;

    float logit[16];
    for (int e = 0; e < 16; ++e) {
        float cl = gate_b[e], nz = noise_b[e];
        for (int c = 0; c < 64; ++c) {
            float f = feat[c];
            cl += f * gate_w[e * 64 + c];
            nz += f * noise_w[e * 64 + c];
        }
        // stable softplus: max(x,0) + log1p(exp(-|x|))
        float sp = fmaxf(nz, 0.0f) + log1pf(expf(-fabsf(nz)));
        logit[e] = cl + noise[b * 16 + e] * (sp + 0.01f);
    }

    // top-2, first-occurrence tie-break (matches lax.top_k ordering)
    int i0 = 0; float l0 = logit[0];
    for (int e = 1; e < 16; ++e)
        if (logit[e] > l0) { l0 = logit[e]; i0 = e; }
    int i1 = -1; float l1 = -INFINITY;
    for (int e = 0; e < 16; ++e) {
        if (e == i0) continue;
        if (logit[e] > l1) { l1 = logit[e]; i1 = e; }
    }
    float e0 = expf(l0 - l0), e1 = expf(l1 - l0);
    float s  = e0 + e1;
    float g0 = e0 / s, g1 = e1 / s;

    for (int e = 0; e < 16; ++e) gsh[b * 16 + e] = 0.0f;
    gsh[b * 16 + i0] = g0;
    gsh[b * 16 + i1] = g1;
    for (int e = 0; e < 16; ++e) out[b * 16 + e] = gsh[b * 16 + e];

    __syncthreads();
    if (b < 16) {                       // deterministic load reduction
        float s2 = 0.0f;
        for (int r = 0; r < 256; ++r) s2 += gsh[r * 16 + b];
        out[4096 + b] = s2;
    }
}

// ---------------------------------------------------------------------------
extern "C" void kernel_launch(void* const* d_in, const int* in_sizes, int n_in,
                              void* d_out, int out_size, void* d_ws, size_t ws_size,
                              hipStream_t stream) {
    const float* x      = (const float*)d_in[0];
    const float* conv_w = (const float*)d_in[1];
    const float* conv_b = (const float*)d_in[2];
    const float* gate_w = (const float*)d_in[3];
    const float* gate_b = (const float*)d_in[4];
    const float* noise_w= (const float*)d_in[5];
    const float* noise_b= (const float*)d_in[6];
    const float* noise  = (const float*)d_in[7];
    float* out = (float*)d_out;

    // Workspace layout: whiT (72 KB) | wloT (72 KB) | partial feats (2 MB)
    unsigned short* whiT = (unsigned short*)d_ws;
    unsigned short* wloT = whiT + 9 * OUT_C * IMG_C;
    float* partial = (float*)(wloT + 9 * OUT_C * IMG_C);

    prep_w_kernel<<<(9 * OUT_C * IMG_C + 255) / 256, 256, 0, stream>>>(
        conv_w, whiT, wloT);
    conv_feat_kernel<<<dim3(RB, 256), 256, 0, stream>>>(
        x, whiT, wloT, conv_b, partial);
    gate_kernel<<<1, 256, 0, stream>>>(
        partial, gate_w, gate_b, noise_w, noise_b, noise, out);
}
